// pred_Encoder_decoder_36910948942058
// MI455X (gfx1250) — matrix operand order repaired and checked by reading, not compiled, not run on hardware
//
#include <hip/hip_runtime.h>
#include <hip/hip_bf16.h>

// ---------------- problem constants ----------------
#define SEQ    128      // encoder length
#define EMB    256
#define HU     512      // H
#define ATTU   512      // ATT
#define VOC    32000
#define DIN    1280     // 2H + EMB
#define KDIM   1024     // 2H
#define XHK    1792     // DIN + H : decoder stacked [x;h] K-dim
#define KT_DEC 56       // XHK/32
#define NT_DEC 128      // 2048/16 n-tiles per direction
#define MAXLEN 128
#define NT_FC  2000     // VOC/16
#define KT_FC  32       // KDIM/32
#define FC_BLOCKS 250   // NT_FC / 8 waves per block

typedef __attribute__((ext_vector_type(16))) __bf16 bf16x16;
typedef __attribute__((ext_vector_type(8)))  float  f32x8;

__device__ __forceinline__ float sigm(float x) { return 1.0f / (1.0f + __expf(-x)); }

__device__ __forceinline__ f32x8 wmma_bf16(bf16x16 a, bf16x16 b, f32x8 c) {
    // emits v_wmma_f32_16x16x32_bf16
    return __builtin_amdgcn_wmma_f32_16x16x32_bf16(false, a, false, b, (short)0, c, false, false);
}

// order-preserving float -> uint map (for 64-bit atomicMax argmax keys)
__device__ __forceinline__ unsigned fkey(float f) {
    unsigned u = __float_as_uint(f);
    return (u & 0x80000000u) ? ~u : (u | 0x80000000u);
}

// ---------------- one-time init ----------------
__global__ void init_k(float* enc_c, int* pred, unsigned long long* gkey, unsigned* gcnt) {
    int i = blockIdx.x * 256 + threadIdx.x;
    if (i < 2 * HU) enc_c[i] = 0.0f;
    if (i == 0) { *pred = 1; *gkey = 0ull; *gcnt = 0u; }   // <SOW>
}

// Pack fcW (KDIM x VOC fp32, row-major) into bf16 WMMA B-fragments.
// Element order: ((nt*KT_FC + kt)*32 + lane)*16 + e
// B 32x16 layout: lanes 0-15 hold K=0..7,16..23 (col n=lane&15), lanes 16-31 hold K=8..15,24..31.
__global__ void pack_fcw(const float* __restrict__ fcW, __bf16* __restrict__ pk) {
    int i = blockIdx.x * 256 + threadIdx.x;
    if (i >= VOC * KDIM) return;
    int e    =  i        & 15;
    int lane = (i >> 4)  & 31;
    int kt   = (i >> 9)  & 31;
    int nt   =  i >> 14;
    int n = nt * 16 + (lane & 15);
    int k = kt * 32 + ((e < 8) ? e : e + 8) + ((lane >= 16) ? 8 : 0);
    pk[i] = (__bf16)fcW[(size_t)k * VOC + n];
}

// Pack decoder [W(1280x2048);U(512x2048)] per direction into bf16 B-fragments.
// Per-dir element order: ((ntl*KT_DEC + kt)*32 + lane)*16 + e ; dir-major.
__global__ void pack_dec(const float* __restrict__ Wf, const float* __restrict__ Uf,
                         const float* __restrict__ Wb, const float* __restrict__ Ub,
                         __bf16* __restrict__ pk) {
    const int dir = blockIdx.y;
    int i = blockIdx.x * 256 + threadIdx.x;                 // 0 .. 1792*2048-1 (exact)
    int e    =  i       & 15;
    int lane = (i >> 4) & 31;
    int kn   =  i >> 9;                                     // ntl*KT_DEC + kt
    int kt   = kn % KT_DEC;
    int ntl  = kn / KT_DEC;
    int n = ntl * 16 + (lane & 15);
    int k = kt * 32 + ((e < 8) ? e : e + 8) + ((lane >= 16) ? 8 : 0);
    const float* W = dir ? Wb : Wf;
    const float* U = dir ? Ub : Uf;
    float v = (k < DIN) ? W[(size_t)k * (4 * HU) + n]
                        : U[(size_t)(k - DIN) * (4 * HU) + n];
    pk[(size_t)dir * (XHK * 4 * HU) + i] = (__bf16)v;
}

// ---------------- encoder LSTM step (fwd+bwd) ----------------
__global__ __launch_bounds__(128)
void enc_step(const int* __restrict__ enc_inp, const float* __restrict__ E_enc,
              const float* __restrict__ Wf, const float* __restrict__ Uf, const float* __restrict__ bfv,
              const float* __restrict__ Wb, const float* __restrict__ Ub, const float* __restrict__ bbv,
              float* __restrict__ enc_out, float* __restrict__ enc_c,
              float* __restrict__ dh0, float* __restrict__ dcf, float* __restrict__ dcb, int t) {
    const int dir = blockIdx.y;
    const int u   = blockIdx.x * 128 + threadIdx.x;          // unit 0..511
    const float* W = dir ? Wb : Wf;
    const float* U = dir ? Ub : Uf;
    const float* B = dir ? bbv : bfv;
    const int s_in  = dir ? (SEQ - 1 - t) : t;               // which token this step consumes
    const int tok   = enc_inp[s_in];
    const float* x  = E_enc + (size_t)tok * EMB;

    float z[4];
#pragma unroll
    for (int g = 0; g < 4; ++g) z[g] = B[g * HU + u];
    for (int i = 0; i < EMB; ++i) {
        float xv = x[i];
        const float* wr = W + (size_t)i * (4 * HU) + u;
#pragma unroll
        for (int g = 0; g < 4; ++g) z[g] += xv * wr[g * HU];
    }
    if (t > 0) {
        const int prev_s = dir ? (SEQ - t) : (t - 1);        // row where previous h was stored
        const float* hp = enc_out + (size_t)prev_s * KDIM + dir * HU;
        for (int j = 0; j < HU; ++j) {
            float hv = hp[j];
            const float* ur = U + (size_t)j * (4 * HU) + u;
#pragma unroll
            for (int g = 0; g < 4; ++g) z[g] += hv * ur[g * HU];
        }
    }
    float c = (t > 0) ? enc_c[dir * HU + u] : 0.0f;
    c = sigm(z[1]) * c + sigm(z[0]) * tanhf(z[2]);
    float h = sigm(z[3]) * tanhf(c);
    enc_c[dir * HU + u] = c;
    enc_out[(size_t)s_in * KDIM + dir * HU + u] = h;
    if (t == SEQ - 1) {                                      // seed decoder states
        dh0[dir * HU + u] = h;
        (dir ? dcb : dcf)[u] = c;
    }
}

// ---------------- enc_proj = enc_out @ W3 + b3 (WMMA GEMM 128x1024x512) ----------------
__global__ __launch_bounds__(256)
void enc_proj_k(const float* __restrict__ enc_out, const float* __restrict__ W3,
                const float* __restrict__ b3, float* __restrict__ enc_proj) {
    const int tid = threadIdx.x, wid = tid >> 5, lane = tid & 31;
    const int gw = blockIdx.x * 8 + wid;                     // 0..255
    const int mt = gw >> 5, nt = gw & 31;                    // 8 x 32 tiles
    const int hi = (lane >= 16) ? 1 : 0;
    const int m = mt * 16 + (lane & 15);
    const int n = nt * 16 + (lane & 15);
    f32x8 c = {0.f, 0.f, 0.f, 0.f, 0.f, 0.f, 0.f, 0.f};
    for (int kt = 0; kt < 32; ++kt) {
        bf16x16 a, b;
#pragma unroll
        for (int e = 0; e < 16; ++e) {
            int ka = kt * 32 + e + hi * 16;                      // A 16x32 bf16 lane layout
            int kb = kt * 32 + e + hi * 16;                      // B 32x16 lane layout
            a[e] = (__bf16)enc_out[(size_t)m * KDIM + ka];
            b[e] = (__bf16)W3[(size_t)kb * ATTU + n];
        }
        c = wmma_bf16(a, b, c);
    }
#pragma unroll
    for (int r = 0; r < 8; ++r) {
        int row = mt * 16 + r + hi * 8;
        int col = nt * 16 + (lane & 15);
        enc_proj[(size_t)row * ATTU + col] = c[r] + b3[col];
    }
}

// ---------------- attention + context + embedding + xh staging (one WG) ----------------
__global__ __launch_bounds__(512)
void attn_k(const float* __restrict__ enc_out, const float* __restrict__ enc_proj,
            const float* __restrict__ W1, const float* __restrict__ b1,
            const float* __restrict__ W2, const float* __restrict__ b2,
            const float* __restrict__ Vw, const float* __restrict__ Vb,
            const float* __restrict__ E_dec, const float* __restrict__ dh_in,
            const int* __restrict__ pred, float* __restrict__ xh,
            float* __restrict__ attn_out) {
    __shared__ float hsh[KDIM];
    __shared__ float q[ATTU];
    __shared__ float sc[SEQ];
    __shared__ float ssum;
    const int tid = threadIdx.x;
    hsh[tid]       = dh_in[tid];
    hsh[tid + 512] = dh_in[tid + 512];
    __syncthreads();
    // q[a] = b1+b2 + dhf@W1 + dhb@W2
    float acc = b1[tid] + b2[tid];
    for (int j = 0; j < HU; ++j) acc += hsh[j] * W1[(size_t)j * ATTU + tid];
    for (int j = 0; j < HU; ++j) acc += hsh[HU + j] * W2[(size_t)j * ATTU + tid];
    q[tid] = acc;
    __syncthreads();
    if (tid < SEQ) {
        float s = Vb[0];
        const float* ep = enc_proj + (size_t)tid * ATTU;
        for (int a = 0; a < ATTU; ++a) s += tanhf(ep[a] + q[a]) * Vw[a];
        sc[tid] = s;
    }
    __syncthreads();
    if (tid == 0) {
        float m = sc[0];
        for (int s = 1; s < SEQ; ++s) m = fmaxf(m, sc[s]);
        float sum = 0.f;
        for (int s = 0; s < SEQ; ++s) { sc[s] = __expf(sc[s] - m); sum += sc[s]; }
        ssum = sum;
    }
    __syncthreads();
    if (tid < SEQ) {
        float a = sc[tid] / ssum;
        sc[tid] = a;
        attn_out[tid] = a;                                  // attns output
    }
    __syncthreads();
    for (int j = tid; j < KDIM; j += 512) {                 // context -> xh[0..1023]
        float cx = 0.f;
        for (int s = 0; s < SEQ; ++s) cx += sc[s] * enc_out[(size_t)s * KDIM + j];
        xh[j] = cx;
    }
    int p = *pred;                                          // previous prediction
    if (tid < EMB) xh[KDIM + tid] = E_dec[(size_t)p * EMB + tid];
    // stage h for the decoder WMMA GEMV: xh = [x(1280); dhf(512); dhb(512)]
    xh[DIN + tid]      = hsh[tid];
    xh[DIN + HU + tid] = hsh[HU + tid];
}

// ---------------- decoder gate pre-activations via WMMA ----------------
// z[dir][col] = bias[col] + [x;h_dir](1792) @ packed[dir](1792x2048)
__global__ __launch_bounds__(256)
void dec_z_k(const bf16x16* __restrict__ Bp, const float* __restrict__ xh,
             const float* __restrict__ bfv, const float* __restrict__ bbv,
             float* __restrict__ zbuf) {
    const int tid = threadIdx.x, wid = tid >> 5, lane = tid & 31;
    const int gw  = blockIdx.x * 8 + wid;                   // 0..255
    const int dir = gw >> 7;
    const int ntl = gw & 127;
    const int hi  = (lane >= 16) ? 1 : 0;
    const bool act = (lane == 0) || (lane == 16);           // only row M=0 is live
    const int hshift = dir ? HU : 0;                        // pick h half of xh
    const bf16x16* bp = Bp + (size_t)dir * (NT_DEC * KT_DEC * 32) + (size_t)ntl * (KT_DEC * 32);
    f32x8 c = {0.f, 0.f, 0.f, 0.f, 0.f, 0.f, 0.f, 0.f};
    for (int kt = 0; kt < KT_DEC; ++kt) {
        bf16x16 a;
        const int base = kt * 32 + hi * 8;
#pragma unroll
        for (int e = 0; e < 16; ++e) {
            int idx = base + ((e < 8) ? e : e + 8);         // A lane K interleave
            int src = (idx >= DIN) ? idx + hshift : idx;    // [x;h_dir]
            a[e] = act ? (__bf16)xh[src] : (__bf16)0.0f;
        }
        bf16x16 b = bp[(size_t)kt * 32 + lane];             // contiguous 32B/lane
        c = wmma_bf16(a, b, c);
    }
    if (lane < 16) {
        int col = ntl * 16 + lane;
        float bias = (dir ? bbv : bfv)[col];
        zbuf[dir * (4 * HU) + col] = c[0] + bias;
    }
}

// ---------------- decoder LSTM state update ----------------
__global__ __launch_bounds__(128)
void dec_update(const float* __restrict__ zbuf, float* __restrict__ dh_out,
                float* __restrict__ dcf, float* __restrict__ dcb) {
    const int dir = blockIdx.y;
    const int u   = blockIdx.x * 128 + threadIdx.x;
    const float* z = zbuf + dir * (4 * HU);
    float zi = z[0 * HU + u], zf = z[1 * HU + u], zg = z[2 * HU + u], zo = z[3 * HU + u];
    float* cb = dir ? dcb : dcf;
    float c = cb[u];
    c = sigm(zf) * c + sigm(zi) * tanhf(zg);
    float hn = sigm(zo) * tanhf(c);
    cb[u] = c;
    dh_out[dir * HU + u] = hn;
}

// ---------------- fc GEMV via WMMA + global atomic argmax ----------------
__global__ __launch_bounds__(256)
void fc_argmax(const bf16x16* __restrict__ Bp, const float* __restrict__ fcb,
               const float* __restrict__ dh, unsigned long long* __restrict__ gkey,
               unsigned* __restrict__ gcnt, int* __restrict__ pred,
               float* __restrict__ out_pred) {
    __shared__ float vals[256];
    __shared__ int   idxs[256];
    const int tid = threadIdx.x, wid = tid >> 5, lane = tid & 31;
    const int nt = blockIdx.x * 8 + wid;                    // 0..1999
    const int hi = (lane >= 16) ? 1 : 0;
    const bool act = (lane == 0) || (lane == 16);           // only row M=0 is live
    f32x8 c = {0.f, 0.f, 0.f, 0.f, 0.f, 0.f, 0.f, 0.f};
    for (int kt = 0; kt < KT_FC; ++kt) {
        bf16x16 a;
        const int base = kt * 32 + hi * 8;
#pragma unroll
        for (int e = 0; e < 16; ++e) {
            int koff = (e < 8) ? e : e + 8;                 // A lane K interleave
            a[e] = act ? (__bf16)dh[base + koff] : (__bf16)0.0f;
        }
        bf16x16 b = Bp[(size_t)(nt * KT_FC + kt) * 32 + lane];  // contiguous 32B/lane
        c = wmma_bf16(a, b, c);
    }
    float v = -3.4e38f;
    int   id = 0x7fffffff;
    if (lane < 16) {                                        // D row 0 lives in lanes 0-15, vgpr 0
        id = nt * 16 + lane;
        v  = c[0] + fcb[id];
    }
    vals[tid] = v;
    idxs[tid] = id;
    __syncthreads();
    if (tid == 0) {
        float bv = vals[0]; int bi = idxs[0];
        for (int j = 1; j < 256; ++j)
            if (vals[j] > bv || (vals[j] == bv && idxs[j] < bi)) { bv = vals[j]; bi = idxs[j]; }
        // pack (value, ~idx) so atomicMax picks max value, lowest index on ties
        unsigned long long key = ((unsigned long long)fkey(bv) << 32) |
                                 (unsigned long long)(~(unsigned)bi);
        atomicMax(gkey, key);
        __threadfence();
        unsigned done = atomicAdd(gcnt, 1u);
        if (done == FC_BLOCKS - 1) {                        // last block: publish + reset
            __threadfence();
            unsigned long long k = atomicMax(gkey, 0ull);
            int idx = (int)(~(unsigned)(k & 0xFFFFFFFFull));
            *pred = idx;
            *out_pred = (float)idx;
            *gkey = 0ull;
            *gcnt = 0u;
        }
    }
}

// ---------------- host ----------------
extern "C" void kernel_launch(void* const* d_in, const int* in_sizes, int n_in,
                              void* d_out, int out_size, void* d_ws, size_t ws_size,
                              hipStream_t stream) {
    (void)in_sizes; (void)n_in; (void)out_size; (void)ws_size;
    const int*   enc_inp = (const int*)  d_in[0];
    const float* E_enc = (const float*)d_in[1];
    const float* eWf = (const float*)d_in[2];
    const float* eUf = (const float*)d_in[3];
    const float* ebf = (const float*)d_in[4];
    const float* eWb = (const float*)d_in[5];
    const float* eUb = (const float*)d_in[6];
    const float* ebb = (const float*)d_in[7];
    const float* E_dec = (const float*)d_in[8];
    const float* dWf = (const float*)d_in[9];
    const float* dUf = (const float*)d_in[10];
    const float* dbf = (const float*)d_in[11];
    const float* dWb = (const float*)d_in[12];
    const float* dUb = (const float*)d_in[13];
    const float* dbb = (const float*)d_in[14];
    const float* W1 = (const float*)d_in[15];
    const float* b1 = (const float*)d_in[16];
    const float* W2 = (const float*)d_in[17];
    const float* b2 = (const float*)d_in[18];
    const float* W3 = (const float*)d_in[19];
    const float* b3 = (const float*)d_in[20];
    const float* Vw = (const float*)d_in[21];
    const float* Vb = (const float*)d_in[22];
    const float* fcW = (const float*)d_in[23];
    const float* fcb = (const float*)d_in[24];
    float* out = (float*)d_out;   // [0..127]=preds, [128 + t*128 + s]=attns

    // workspace carve-out (~81 MB: packed bf16 fcW 64MB + packed dec 14.7MB + state)
    char* wsb = (char*)d_ws;
    size_t off = 0;
    auto take = [&](size_t bytes) -> char* {
        char* p = wsb + off;
        off += (bytes + 255) & ~(size_t)255;
        return p;
    };
    __bf16* fcw_pk  = (__bf16*)take((size_t)VOC * KDIM * 2);
    __bf16* dec_pk  = (__bf16*)take((size_t)2 * XHK * 4 * HU * 2);
    float*  enc_out = (float*) take((size_t)SEQ * KDIM * 4);
    float*  enc_pr  = (float*) take((size_t)SEQ * ATTU * 4);
    float*  enc_c   = (float*) take(2 * HU * 4);
    float*  dh0     = (float*) take(KDIM * 4);
    float*  dh1     = (float*) take(KDIM * 4);
    float*  dcf     = (float*) take(HU * 4);
    float*  dcb     = (float*) take(HU * 4);
    float*  xh      = (float*) take((DIN + KDIM) * 4);      // [x(1280); dhf; dhb]
    float*  zbuf    = (float*) take(2 * 4 * HU * 4);
    unsigned long long* gkey = (unsigned long long*)take(256);
    unsigned*           gcnt = (unsigned*)take(256);
    int*                pred = (int*)take(256);

    init_k<<<4, 256, 0, stream>>>(enc_c, pred, gkey, gcnt);
    pack_fcw<<<(VOC * KDIM) / 256, 256, 0, stream>>>(fcW, fcw_pk);
    pack_dec<<<dim3((XHK * 4 * HU) / 256, 2), 256, 0, stream>>>(dWf, dUf, dWb, dUb, dec_pk);

    for (int t = 0; t < SEQ; ++t)
        enc_step<<<dim3(4, 2), 128, 0, stream>>>(enc_inp, E_enc, eWf, eUf, ebf,
                                                 eWb, eUb, ebb, enc_out, enc_c,
                                                 dh0, dcf, dcb, t);

    enc_proj_k<<<32, 256, 0, stream>>>(enc_out, W3, b3, enc_pr);

    for (int t = 0; t < MAXLEN; ++t) {
        float* dh_in  = (t & 1) ? dh1 : dh0;
        float* dh_out = (t & 1) ? dh0 : dh1;
        attn_k<<<1, 512, 0, stream>>>(enc_out, enc_pr, W1, b1, W2, b2, Vw, Vb,
                                      E_dec, dh_in, pred, xh, out + 128 + t * 128);
        dec_z_k<<<32, 256, 0, stream>>>((const bf16x16*)dec_pk, xh, dbf, dbb, zbuf);
        dec_update<<<dim3(4, 2), 128, 0, stream>>>(zbuf, dh_out, dcf, dcb);
        fc_argmax<<<FC_BLOCKS, 256, 0, stream>>>((const bf16x16*)fcw_pk, fcb, dh_out,
                                                 gkey, gcnt, pred, out + t);
    }
}
